// MaskedGraphAF_55155970015788
// MI455X (gfx1250) — matrix operand order repaired and checked by reading, not compiled, not run on hardware
//
#include <hip/hip_runtime.h>
#include <hip/hip_bf16.h>
#include <math.h>

typedef __attribute__((ext_vector_type(16))) __bf16 v16bf;
typedef __attribute__((ext_vector_type(8)))  float  v8f;

#define B_   64
#define N_   384
#define E_   3
#define NH_  128
#define ROWS (B_ * N_)   // 24576

static __device__ __forceinline__ v8f wmma_bf16(v16bf a, v16bf b, v8f c) {
    return __builtin_amdgcn_wmma_f32_16x16x32_bf16(false, a, false, b, (short)0, c, false, false);
}

// ---------------------------------------------------------------- weight convert + transpose
// dst[e][j][k] = (bf16) src[e][k][j]   (src: (E, CIN, 128), dst: (E, 128, CIN))
__global__ void cvt_transpose_kernel(const float* __restrict__ src,
                                     __bf16* __restrict__ dst, int cin) {
    int idx = blockIdx.x * blockDim.x + threadIdx.x;
    int tot = E_ * NH_ * cin;
    if (idx >= tot) return;
    int e = idx / (NH_ * cin);
    int r = idx - e * (NH_ * cin);
    int j = r / cin;          // cout
    int k = r - j * cin;      // cin
    dst[idx] = (__bf16)src[(size_t)e * cin * NH_ + (size_t)k * NH_ + j];
}

// ---------------------------------------------------------------- embed: h0 = x @ emb_w^T  (bf16 out)
__global__ void embed_kernel(const float* __restrict__ x,
                             const float* __restrict__ emb_w,
                             __bf16* __restrict__ h0) {
    __shared__ float sw[16 * 16];
    int t = threadIdx.x;
    sw[t & 255] = emb_w[t & 255];
    __syncthreads();
    int row = blockIdx.x * blockDim.x + t;
    const float* xp = x + (size_t)row * 16;
    float xin[16];
#pragma unroll
    for (int d = 0; d < 16; d++) xin[d] = xp[d];
    union { uint4 u[2]; __bf16 h[16]; } o;
#pragma unroll
    for (int j = 0; j < 16; j++) {
        float acc = 0.f;
#pragma unroll
        for (int d = 0; d < 16; d++) acc += xin[d] * sw[j * 16 + d];
        o.h[j] = (__bf16)acc;
    }
    uint4* dp = reinterpret_cast<uint4*>(h0 + (size_t)row * 16);
    dp[0] = o.u[0]; dp[1] = o.u[1];
}

// ---------------------------------------------------------------- support: sT[b,e,cout,node] = (h[b] @ W[e])^T
// h: (B*N, CIN) bf16, WT: (E, 128, CIN) bf16 (transposed!), sT: (B*E, 128, 384) bf16
template<int CIN>
__global__ void support_kernel(const __bf16* __restrict__ h,
                               const __bf16* __restrict__ WT,
                               __bf16* __restrict__ sT) {
    const int lane = threadIdx.x & 31;
    const int wid  = blockIdx.x * (blockDim.x >> 5) + (threadIdx.x >> 5);
    const int tn   = wid & 7;
    const int tm   = (wid >> 3) % 24;
    const int be   = wid / (8 * 24);
    const int b    = be / E_;
    const int e    = be % E_;
    const int half = lane >> 4;
    const int l    = lane & 15;
    const int m    = tm * 16 + l;
    const int n    = tn * 16 + l;

    const __bf16* hrow = h + (size_t)(b * N_ + m) * CIN;
    const __bf16* wrow = WT + ((size_t)e * NH_ + n) * CIN;

    v8f c = {};
    for (int k0 = 0; k0 < CIN; k0 += 32) {
        v16bf a;
#pragma unroll
        for (int g = 0; g < 2; g++) {
            int kb = k0 + 16 * half + 8 * g;
            if (kb < CIN) {
                union { uint4 u; __bf16 hh[8]; } t;
                t.u = *reinterpret_cast<const uint4*>(hrow + kb);
#pragma unroll
                for (int i = 0; i < 8; i++) a[g * 8 + i] = t.hh[i];
            } else {
#pragma unroll
                for (int i = 0; i < 8; i++) a[g * 8 + i] = (__bf16)0.f;
            }
        }
        v16bf bb;
        int kb = k0 + 16 * half;
        if (kb < CIN) {
            union { uint4 u[2]; __bf16 hh[16]; } t;
            const uint4* p = reinterpret_cast<const uint4*>(wrow + kb);
            t.u[0] = p[0]; t.u[1] = p[1];
#pragma unroll
            for (int i = 0; i < 16; i++) bb[i] = t.hh[i];
        } else {
#pragma unroll
            for (int i = 0; i < 16; i++) bb[i] = (__bf16)0.f;
        }
        c = wmma_bf16(a, bb, c);
    }
    union { uint4 u; __bf16 hh[8]; } o;
#pragma unroll
    for (int r = 0; r < 8; r++) o.hh[r] = (__bf16)c[r];
    int nodebase = tm * 16 + 8 * half;
    *reinterpret_cast<uint4*>(sT + ((size_t)be * NH_ + n) * N_ + nodebase) = o.u;
}

// ---------------------------------------------------------------- out[b] = sum_e relu?(adj[b,e] @ support[b,e])
// Each block: fixed b, one 64-cout half, 8 m-tiles (one per wave), wave tile = 16x64.
// Support panel (64 x 384 bf16 = 48 KB) staged in LDS via async copy, double buffered over e.
template<bool RELU, bool OUTBF>
__global__ __launch_bounds__(256) void adjgemm_kernel(const float* __restrict__ adj,
                                                      const __bf16* __restrict__ sT,
                                                      void* __restrict__ outp) {
    constexpr int ROWB  = 768;           // bytes per panel row in memory (384 bf16)
    constexpr int ROWS_ = 784;           // padded LDS row stride (+16 B: kills 192-dword bank alignment)
    constexpr int PANEL = 64 * ROWS_;    // 50176 B
    __shared__ __align__(16) char sbuf[2 * PANEL];   // ~98 KB of 320 KB WGP LDS

    const int blk  = blockIdx.x;         // B_ * 2 * 3 blocks
    const int nh   = blk & 1;
    const int mg   = (blk >> 1) % 3;
    const int b    = blk / 6;

    const int t    = threadIdx.x;
    const int lane = t & 31;
    const int wv   = t >> 5;             // 0..7
    const int half = lane >> 4;
    const int l    = lane & 15;

    const int tm    = mg * 8 + wv;       // 0..23
    const int m     = tm * 16 + l;
    const int cout0 = nh * 64;

    auto load_panel = [&](int e, int bi) {
        const char* src = reinterpret_cast<const char*>(
            sT + ((size_t)(b * E_ + e) * NH_ + cout0) * N_);
        uint32_t ldsbase = (uint32_t)(size_t)(void*)(sbuf + bi * PANEL);
        // 64 rows x 48 chunks of 16 B; 3072 chunks over 256 threads
        for (int c = t; c < 64 * 48; c += 256) {
            int row = c / 48, col = c - row * 48;
            uint32_t go = (uint32_t)(row * ROWB + col * 16);
            uint32_t lo = ldsbase + (uint32_t)(row * ROWS_ + col * 16);
            asm volatile("global_load_async_to_lds_b128 %0, %1, %2"
                         :: "v"(lo), "v"(go), "s"(src) : "memory");
        }
    };

    v8f ctot[4] = {};

    load_panel(0, 0);
    asm volatile("s_wait_asynccnt 0x0" ::: "memory");
    __syncthreads();

    for (int e = 0; e < E_; e++) {
        if (e + 1 < E_) load_panel(e + 1, (e + 1) & 1);

        const char*  bufe = sbuf + (e & 1) * PANEL;
        const float* arow = adj + ((size_t)(b * E_ + e) * N_ + m) * N_;
        v8f ce[4] = {};
        for (int k0 = 0; k0 < N_; k0 += 32) {
            if (k0 + 32 < N_)
                __builtin_prefetch(arow + k0 + 32 + 8 * half, 0, 0);
            v16bf a;
#pragma unroll
            for (int g = 0; g < 2; g++) {
                union { uint4 u[2]; float f[8]; } tt;
                const uint4* p = reinterpret_cast<const uint4*>(arow + k0 + 16 * half + 8 * g);
                tt.u[0] = p[0]; tt.u[1] = p[1];
#pragma unroll
                for (int i = 0; i < 8; i++) a[g * 8 + i] = (__bf16)tt.f[i];
            }
#pragma unroll
            for (int s = 0; s < 4; s++) {
                int cl = s * 16 + l;                 // cout_local 0..63 (lane's B column)
                union { uint4 u[2]; __bf16 hh[16]; } tt;
                const uint4* p = reinterpret_cast<const uint4*>(
                    bufe + cl * ROWS_ + (k0 + 16 * half) * 2);
                tt.u[0] = p[0]; tt.u[1] = p[1];
                v16bf bb;
#pragma unroll
                for (int i = 0; i < 16; i++) bb[i] = tt.hh[i];
                ce[s] = wmma_bf16(a, bb, ce[s]);
            }
        }
#pragma unroll
        for (int s = 0; s < 4; s++)
#pragma unroll
            for (int r = 0; r < 8; r++)
                ctot[s][r] += RELU ? fmaxf(ce[s][r], 0.f) : ce[s][r];

        if (e + 1 < E_) {
            asm volatile("s_wait_asynccnt 0x0" ::: "memory");
            __syncthreads();
        }
    }

    if (OUTBF) {
        __bf16* ho = reinterpret_cast<__bf16*>(outp);
#pragma unroll
        for (int s = 0; s < 4; s++) {
            int n = cout0 + s * 16 + l;
#pragma unroll
            for (int r = 0; r < 8; r++) {
                int mr = tm * 16 + 8 * half + r;
                ho[(size_t)(b * N_ + mr) * NH_ + n] = (__bf16)ctot[s][r];
            }
        }
    } else {
        float* ho = reinterpret_cast<float*>(outp);
#pragma unroll
        for (int s = 0; s < 4; s++) {
            int n = cout0 + s * 16 + l;
#pragma unroll
            for (int r = 0; r < 8; r++) {
                int mr = tm * 16 + 8 * half + r;
                ho[(size_t)(b * N_ + mr) * NH_ + n] = ctot[s][r];
            }
        }
    }
}

// ---------------------------------------------------------------- stat head (VALU)
__global__ void stat_kernel(const float* __restrict__ h3,
                            const float* __restrict__ w1, const float* __restrict__ b1,
                            const float* __restrict__ w2, const float* __restrict__ b2,
                            const float* __restrict__ rescale,
                            float* __restrict__ out) {
    __shared__ float sw1[64 * 128];
    __shared__ float sw2[32 * 64];
    __shared__ float sb1[64];
    __shared__ float sb2[32];
    for (int i = threadIdx.x; i < 64 * 128; i += blockDim.x) sw1[i] = w1[i];
    for (int i = threadIdx.x; i < 32 * 64;  i += blockDim.x) sw2[i] = w2[i];
    if (threadIdx.x < 64) sb1[threadIdx.x] = b1[threadIdx.x];
    if (threadIdx.x < 32) sb2[threadIdx.x] = b2[threadIdx.x];
    __syncthreads();

    int row = blockIdx.x * blockDim.x + threadIdx.x;
    const float* hr = h3 + (size_t)row * 128;
    float hid[64];
#pragma unroll
    for (int j = 0; j < 64; j++) hid[j] = sb1[j];
    for (int d = 0; d < 128; d++) {
        float xv = hr[d];
#pragma unroll
        for (int j = 0; j < 64; j++) hid[j] += xv * sw1[j * 128 + d];
    }
#pragma unroll
    for (int j = 0; j < 64; j++) hid[j] = tanhf(hid[j]);
    float sc = expf(rescale[0]);
    for (int o = 0; o < 32; o++) {
        float acc = sb2[o];
#pragma unroll
        for (int j = 0; j < 64; j++) acc += hid[j] * sw2[o * 64 + j];
        if (o < 16) out[(size_t)row * 16 + o] = sc * tanhf(acc);
        else        out[(size_t)ROWS * 16 + (size_t)row * 16 + (o - 16)] = acc;
    }
}

// ---------------------------------------------------------------- launcher
extern "C" void kernel_launch(void* const* d_in, const int* in_sizes, int n_in,
                              void* d_out, int out_size, void* d_ws, size_t ws_size,
                              hipStream_t stream) {
    const float* x     = (const float*)d_in[0];
    const float* adj   = (const float*)d_in[1];
    const float* emb_w = (const float*)d_in[2];
    const float* gc1_w = (const float*)d_in[3];
    const float* gc2_w = (const float*)d_in[4];
    const float* gc3_w = (const float*)d_in[5];
    const float* st_w1 = (const float*)d_in[6];
    const float* st_b1 = (const float*)d_in[7];
    const float* st_w2 = (const float*)d_in[8];
    const float* st_b2 = (const float*)d_in[9];
    const float* resc  = (const float*)d_in[10];

    char* ws = (char*)d_ws;
    size_t off = 0;
    auto take = [&](size_t bytes) -> char* {
        char* p = ws + off;
        off = (off + bytes + 255) & ~(size_t)255;
        return p;
    };
    __bf16* h0  = (__bf16*)take((size_t)ROWS * 16 * 2);
    __bf16* h1  = (__bf16*)take((size_t)ROWS * NH_ * 2);
    __bf16* h2  = (__bf16*)take((size_t)ROWS * NH_ * 2);
    __bf16* sT  = (__bf16*)take((size_t)B_ * E_ * NH_ * N_ * 2);
    float*  h3  = (float* )take((size_t)ROWS * NH_ * 4);
    __bf16* w1t = (__bf16*)take((size_t)E_ * NH_ * 16 * 2);
    __bf16* w2t = (__bf16*)take((size_t)E_ * NH_ * NH_ * 2);
    __bf16* w3t = (__bf16*)take((size_t)E_ * NH_ * NH_ * 2);

    cvt_transpose_kernel<<<(E_ * NH_ * 16  + 255) / 256, 256, 0, stream>>>(gc1_w, w1t, 16);
    cvt_transpose_kernel<<<(E_ * NH_ * NH_ + 255) / 256, 256, 0, stream>>>(gc2_w, w2t, NH_);
    cvt_transpose_kernel<<<(E_ * NH_ * NH_ + 255) / 256, 256, 0, stream>>>(gc3_w, w3t, NH_);
    embed_kernel<<<ROWS / 256, 256, 0, stream>>>(x, emb_w, h0);

    // support: 192 be * 24 * 8 = 36864 wave-tiles / 8 per block
    // adjgemm: 64 b * 2 cout-halves * 3 m-groups = 384 blocks (8 waves each, 16x64 per wave)
    support_kernel<16 ><<<4608, 256, 0, stream>>>(h0, w1t, sT);
    adjgemm_kernel<true,  true ><<<B_ * 6, 256, 0, stream>>>(adj, sT, (void*)h1);

    support_kernel<128><<<4608, 256, 0, stream>>>(h1, w2t, sT);
    adjgemm_kernel<true,  true ><<<B_ * 6, 256, 0, stream>>>(adj, sT, (void*)h2);

    support_kernel<128><<<4608, 256, 0, stream>>>(h2, w3t, sT);
    adjgemm_kernel<false, false><<<B_ * 6, 256, 0, stream>>>(adj, sT, (void*)h3);

    stat_kernel<<<ROWS / 128, 128, 0, stream>>>(h3, st_w1, st_b1, st_w2, st_b2, resc, (float*)d_out);
}
